// SoftGroupModel_72928544686197
// MI455X (gfx1250) — matrix operand order repaired and checked by reading, not compiled, not run on hardware
//
#include <hip/hip_runtime.h>

// ---------------------------------------------------------------------------
// SoftGroup sparse-conv U-Net on gfx1250 (MI455X).
// f16 feature maps end-to-end; gather staged into LDS with
// GLOBAL_LOAD_ASYNC_TO_LDS_B128 (double buffered, s_wait_asynccnt) and
// consumed by v_wmma_f32_16x16x32_f16. Weights pre-transposed to
// (27, Cout, Cin_pad) f16 so each B-lane loads 32 contiguous bytes.
// ---------------------------------------------------------------------------

typedef _Float16 half_t;
typedef half_t v16h __attribute__((ext_vector_type(16)));
typedef float  v8f  __attribute__((ext_vector_type(8)));
typedef int    v4i  __attribute__((ext_vector_type(4)));

// ---- CDNA5 async-copy helpers (guarded; fallback keeps compile clean) -----
__device__ __forceinline__ void async_copy16(const half_t* g, half_t* l) {
#if __has_builtin(__builtin_amdgcn_global_load_async_to_lds_b128)
  typedef __attribute__((address_space(1))) v4i gv4i;
  typedef __attribute__((address_space(3))) v4i lv4i;
  __builtin_amdgcn_global_load_async_to_lds_b128((gv4i*)g, (lv4i*)l, 0, 0);
#else
  *(uint4*)l = *(const uint4*)g;
#endif
}

__device__ __forceinline__ void async_wait0() {
#if __has_builtin(__builtin_amdgcn_s_wait_asynccnt)
  __builtin_amdgcn_s_wait_asynccnt(0);
#else
  asm volatile("s_wait_asynccnt 0" ::: "memory");
#endif
}

// ---------------------------- utility kernels ------------------------------

__global__ void zero_f32(float* __restrict__ p, long n) {
  long i = (long)blockIdx.x * blockDim.x + threadIdx.x;
  if (i < n) p[i] = 0.f;
}

__global__ void zero_f16(half_t* __restrict__ p, int n) {
  int i = blockIdx.x * blockDim.x + threadIdx.x;
  if (i < n) p[i] = (half_t)0.f;
}

__global__ void vox_accum(const float* __restrict__ feat, const int* __restrict__ pvid,
                          float* __restrict__ vox, float* __restrict__ cnt, int npts) {
  int p = blockIdx.x * blockDim.x + threadIdx.x;
  if (p >= npts) return;
  int v = pvid[p];
  atomicAdd(&cnt[v], 1.f);
#pragma unroll
  for (int j = 0; j < 9; ++j) atomicAdd(&vox[(long)v * 9 + j], feat[(long)p * 9 + j]);
}

// f32 voxel sums -> f16 padded feature rows (stride 32, 9 valid ch, pad row 0)
__global__ void vox_to_f16(const float* __restrict__ voxf, const float* __restrict__ cnt,
                           half_t* __restrict__ out, int nvox) {
  long i = (long)blockIdx.x * blockDim.x + threadIdx.x;
  if (i >= (long)(nvox + 1) * 32) return;
  int r = (int)(i >> 5), ci = (int)(i & 31);
  float v = 0.f;
  if (r < nvox && ci < 9) {
    float c = cnt[r];
    v = voxf[(long)r * 9 + ci] / (c < 1.f ? 1.f : c);
  }
  out[i] = (half_t)v;
}

// W (27, Cin, Cout) f32 -> (27, Cout, Cin_pad) f16, zero-padded K tail
__global__ void wcvt(const float* __restrict__ W, half_t* __restrict__ out,
                     int cin, int cinp, int cout) {
  long n = 27l * cout * cinp;
  long e = (long)blockIdx.x * blockDim.x + threadIdx.x;
  if (e >= n) return;
  int ci = (int)(e % cinp);
  long t = e / cinp;
  int co = (int)(t % cout);
  int k  = (int)(t / cout);
  float v = (ci < cin) ? W[((long)k * cin + ci) * cout + co] : 0.f;
  out[e] = (half_t)v;
}

__global__ void concat2_f16(const half_t* __restrict__ A, int ca,
                            const half_t* __restrict__ B, int cb,
                            half_t* __restrict__ out, int nrows /* incl pad row */) {
  int ct = ca + cb;
  long i = (long)blockIdx.x * blockDim.x + threadIdx.x;
  if (i >= (long)nrows * ct) return;
  int r = (int)(i / ct), c = (int)(i - (long)r * ct);
  out[i] = (c < ca) ? A[(long)r * ca + c] : B[(long)r * cb + (c - ca)];
}

// ------------------------ sparse conv, WMMA core ---------------------------
// out[o,:] = ReLU((sum_k feat[nbr[o,k]] @ W[k]) * g + b), f16 in / f16 out.
// feat has n_in+1 rows (row n_in is the zero pad row the rulebook points at).
template <int CINP>
__global__ void __launch_bounds__(128)
spconv_wmma(const half_t* __restrict__ feat, const int* __restrict__ nbr,
            const half_t* __restrict__ wt, const float* __restrict__ g,
            const float* __restrict__ b, half_t* __restrict__ out,
            int n_in, int n_out, int cout) {
  constexpr int KCH = CINP / 32;        // K chunks of 32
  constexpr int CPR = CINP / 8;         // 16-byte chunks per row
  constexpr int CHUNKS = 16 * CPR;      // chunks per 16-row stage
  __shared__ half_t sA[2][16 * CINP];   // double-buffered gather tile

  const int tid  = threadIdx.x;
  const int lane = tid & 31;
  const int wave = tid >> 5;
  const int tile0   = blockIdx.x * 16;
  const int colbase = blockIdx.y * 64 + wave * 16;
  const int m  = lane & 15;             // A-row / B-col / D-col in tile
  const int hh = lane >> 4;             // lane-half select per ISA layouts

  // issue async gather of the 16 rows for offset k into buffer `buf`
  auto stage = [&](int k, int buf) {
    for (int c = tid; c < CHUNKS; c += 128) {
      int r = c / CPR;                  // power-of-two -> shift
      int e = (c - r * CPR) * 8;        // half-element offset in row
      int o = tile0 + r;
      int idx = (o < n_out) ? nbr[(long)o * 27 + k] : n_in;  // pad row
      async_copy16(feat + (long)idx * CINP + e, &sA[buf][r * CINP + e]);
    }
  };

  v8f acc = {};
  stage(0, 0);

  for (int k = 0; k < 27; ++k) {
    async_wait0();        // this wave's copies for buffer k&1 have landed
    __syncthreads();      // all waves' copies landed; k-1 reads all done
    if (k + 1 < 27) stage(k + 1, (k + 1) & 1);  // overlap DMA with WMMA

    if (colbase < cout) {
      if (k + 1 < 27)
        __builtin_prefetch(&wt[((long)(k + 1) * cout + colbase + m) * CINP], 0, 1);
      const half_t* __restrict__ arow = &sA[k & 1][m * CINP];
      const half_t* __restrict__ wrow = &wt[((long)k * cout + colbase + m) * CINP];
#pragma unroll
      for (int kc = 0; kc < KCH; ++kc) {
        // 16-bit A 16x32 layout: lane m = row m; VGPR v -> K pairs
        // K = 2v + 16*hh (matches B's contiguous k order)
        union { v16h h; unsigned u[8]; } A;
#pragma unroll
        for (int v2 = 0; v2 < 8; ++v2) {
          int Kb = 2 * v2 + 16 * hh + kc * 32;
          A.u[v2] = *(const unsigned*)&arow[Kb];   // ds_load_b32
        }
        // 16-bit B 32x16 layout: lane m = col m; element i -> K = i + 16*hh
        v16h B = *(const v16h*)(wrow + kc * 32 + 16 * hh);  // 2x b128
        acc = __builtin_amdgcn_wmma_f32_16x16x32_f16(
            false, A.h, false, B, (short)0, acc, false, false);
      }
    }
  }

  // f32 C/D layout: VGPR i -> M = i + 8*hh, N = lane&15
  if (colbase < cout) {
    int co = colbase + m;
    float gg = g[co], bb = b[co];
#pragma unroll
    for (int i = 0; i < 8; ++i) {
      int o = tile0 + i + 8 * hh;
      if (o < n_out) {
        float v = acc[i] * gg + bb;
        out[(long)o * cout + co] = (half_t)(v > 0.f ? v : 0.f);
      }
    }
  }
}

// ------------------------- devoxelize + MLP heads --------------------------
__global__ void heads_kernel(const half_t* __restrict__ vf, const int* __restrict__ pvid,
                             const float* __restrict__ sW1, const float* __restrict__ sW2,
                             const float* __restrict__ sb1, const float* __restrict__ sb2,
                             const float* __restrict__ oW1, const float* __restrict__ oW2,
                             const float* __restrict__ ob1, const float* __restrict__ ob2,
                             float* __restrict__ out_pf, float* __restrict__ out_sem,
                             float* __restrict__ out_off, int npts) {
  int p = blockIdx.x * blockDim.x + threadIdx.x;
  if (p >= npts) return;
  int v = pvid[p];
  float f[32], h[32];
#pragma unroll
  for (int i = 0; i < 32; ++i) {
    f[i] = (float)vf[(long)v * 32 + i];
    out_pf[(long)p * 32 + i] = f[i];
  }
  for (int j = 0; j < 32; ++j) {
    float s = sb1[j];
    for (int i = 0; i < 32; ++i) s += f[i] * sW1[i * 32 + j];
    h[j] = s > 0.f ? s : 0.f;
  }
  for (int j = 0; j < 2; ++j) {
    float s = sb2[j];
    for (int i = 0; i < 32; ++i) s += h[i] * sW2[i * 2 + j];
    out_sem[(long)p * 2 + j] = s;
  }
  for (int j = 0; j < 32; ++j) {
    float s = ob1[j];
    for (int i = 0; i < 32; ++i) s += f[i] * oW1[i * 32 + j];
    h[j] = s > 0.f ? s : 0.f;
  }
  for (int j = 0; j < 3; ++j) {
    float s = ob2[j];
    for (int i = 0; i < 32; ++i) s += h[i] * oW2[i * 3 + j];
    out_off[(long)p * 3 + j] = s;
  }
}

// ------------------------------- host side ---------------------------------
// Input index map: setup_inputs() dict order; 'params' pytree flattened with
// sorted dict keys (jax convention): cbr {W,b,g}, enc {W1,W2,b1,b2,g1,g2},
// head {W1,W2,b1,b2}.
enum {
  IN_FEAT = 0,
  P_DEC1_W = 1,  P_DEC1_B, P_DEC1_G,
  P_DEC2_W = 4,  P_DEC2_B, P_DEC2_G,
  P_DEC3_W = 7,  P_DEC3_B, P_DEC3_G,
  P_DN1_W = 10,  P_DN1_B, P_DN1_G,
  P_DN2_W = 13,  P_DN2_B, P_DN2_G,
  P_DN3_W = 16,  P_DN3_B, P_DN3_G,
  P_E0_W1 = 19,  P_E0_W2, P_E0_B1, P_E0_B2, P_E0_G1, P_E0_G2,
  P_E1_W1 = 25,  P_E1_W2, P_E1_B1, P_E1_B2, P_E1_G1, P_E1_G2,
  P_E2_W1 = 31,  P_E2_W2, P_E2_B1, P_E2_B2, P_E2_G1, P_E2_G2,
  P_E3_W1 = 37,  P_E3_W2, P_E3_B1, P_E3_B2, P_E3_G1, P_E3_G2,
  P_INP_W = 43,  P_INP_B, P_INP_G,
  P_OFF_W1 = 46, P_OFF_W2, P_OFF_B1, P_OFF_B2,
  P_SEM_W1 = 50, P_SEM_W2, P_SEM_B1, P_SEM_B2,
  P_UP1_W = 54,  P_UP1_B, P_UP1_G,
  P_UP2_W = 57,  P_UP2_B, P_UP2_G,
  P_UP3_W = 60,  P_UP3_B, P_UP3_G,
  IN_PVID = 63,
  IN_NBR0 = 64, IN_NBR1, IN_NBR2, IN_NBR3,
  IN_DN1 = 68, IN_DN2, IN_DN3,
  IN_INV1 = 71, IN_INV2, IN_INV3,
};

struct LayerW { int w, b, g, cin, cinp, cout; };
static const LayerW kLayers[18] = {
  {P_INP_W,  P_INP_B,  P_INP_G,    9,  32,  32},  // 0 inp
  {P_E0_W1,  P_E0_B1,  P_E0_G1,   32,  32,  32},  // 1 enc0a
  {P_E0_W2,  P_E0_B2,  P_E0_G2,   32,  32,  32},  // 2 enc0b
  {P_DN1_W,  P_DN1_B,  P_DN1_G,   32,  32,  64},  // 3 down1
  {P_E1_W1,  P_E1_B1,  P_E1_G1,   64,  64,  64},  // 4 enc1a
  {P_E1_W2,  P_E1_B2,  P_E1_G2,   64,  64,  64},  // 5 enc1b
  {P_DN2_W,  P_DN2_B,  P_DN2_G,   64,  64, 128},  // 6 down2
  {P_E2_W1,  P_E2_B1,  P_E2_G1,  128, 128, 128},  // 7 enc2a
  {P_E2_W2,  P_E2_B2,  P_E2_G2,  128, 128, 128},  // 8 enc2b
  {P_DN3_W,  P_DN3_B,  P_DN3_G,  128, 128, 256},  // 9 down3
  {P_E3_W1,  P_E3_B1,  P_E3_G1,  256, 256, 256},  // 10 enc3a
  {P_E3_W2,  P_E3_B2,  P_E3_G2,  256, 256, 256},  // 11 enc3b
  {P_UP3_W,  P_UP3_B,  P_UP3_G,  256, 256, 128},  // 12 up3
  {P_DEC3_W, P_DEC3_B, P_DEC3_G, 256, 256, 128},  // 13 dec3
  {P_UP2_W,  P_UP2_B,  P_UP2_G,  128, 128,  64},  // 14 up2
  {P_DEC2_W, P_DEC2_B, P_DEC2_G, 128, 128,  64},  // 15 dec2
  {P_UP1_W,  P_UP1_B,  P_UP1_G,   64,  64,  32},  // 16 up1
  {P_DEC1_W, P_DEC1_B, P_DEC1_G,  64,  64,  32},  // 17 dec1
};

static inline size_t align256(size_t x) { return (x + 255) & ~(size_t)255; }

static void launch_spconv(hipStream_t s, const half_t* fin, const int* nbr,
                          const half_t* wt, const float* g, const float* b,
                          half_t* out, int n_in, int n_out, int cinp, int cout) {
  dim3 grid((n_out + 15) / 16, (cout + 63) / 64);
  switch (cinp) {
    case 32:  spconv_wmma<32><<<grid, 128, 0, s>>>(fin, nbr, wt, g, b, out, n_in, n_out, cout); break;
    case 64:  spconv_wmma<64><<<grid, 128, 0, s>>>(fin, nbr, wt, g, b, out, n_in, n_out, cout); break;
    case 128: spconv_wmma<128><<<grid, 128, 0, s>>>(fin, nbr, wt, g, b, out, n_in, n_out, cout); break;
    default:  spconv_wmma<256><<<grid, 128, 0, s>>>(fin, nbr, wt, g, b, out, n_in, n_out, cout); break;
  }
}

extern "C" void kernel_launch(void* const* d_in, const int* in_sizes, int n_in,
                              void* d_out, int out_size, void* d_ws, size_t ws_size,
                              hipStream_t stream) {
  const float* feat = (const float*)d_in[IN_FEAT];
  const int* pvid = (const int*)d_in[IN_PVID];
  const int* nbr0 = (const int*)d_in[IN_NBR0];
  const int* nbr1 = (const int*)d_in[IN_NBR1];
  const int* nbr2 = (const int*)d_in[IN_NBR2];
  const int* nbr3 = (const int*)d_in[IN_NBR3];
  const int* dn1 = (const int*)d_in[IN_DN1];
  const int* dn2 = (const int*)d_in[IN_DN2];
  const int* dn3 = (const int*)d_in[IN_DN3];
  const int* iv1 = (const int*)d_in[IN_INV1];
  const int* iv2 = (const int*)d_in[IN_INV2];
  const int* iv3 = (const int*)d_in[IN_INV3];

  const int npts = in_sizes[IN_FEAT] / 9;
  const int n0 = in_sizes[IN_NBR0] / 27;
  const int n1 = in_sizes[IN_NBR1] / 27;
  const int n2 = in_sizes[IN_NBR2] / 27;
  const int n3 = in_sizes[IN_NBR3] / 27;

  // ---- workspace bump allocator ----
  char* ws = (char*)d_ws;
  size_t off = 0;
  auto allocF = [&](size_t elems) -> float* {
    float* p = (float*)(ws + off);
    off = align256(off + elems * sizeof(float));
    return p;
  };
  auto allocH = [&](size_t elems) -> half_t* {
    half_t* p = (half_t*)(ws + off);
    off = align256(off + elems * sizeof(half_t));
    return p;
  };

  // f16 transposed weights
  half_t* wt16[18];
  for (int i = 0; i < 18; ++i) {
    const LayerW& L = kLayers[i];
    size_t ne = 27ul * L.cout * L.cinp;
    wt16[i] = allocH(ne);
    wcvt<<<(unsigned)((ne + 255) / 256), 256, 0, stream>>>(
        (const float*)d_in[L.w], wt16[i], L.cin, L.cinp, L.cout);
  }

  // f16 feature buffers, each with one extra zero pad row (index n)
  float* cnt  = allocF(n0);
  float* voxf = allocF((size_t)n0 * 9);
  half_t* vox = allocH((size_t)(n0 + 1) * 32);
  half_t* x0  = allocH((size_t)(n0 + 1) * 32);
  half_t* t0  = allocH((size_t)(n0 + 1) * 32);
  half_t* e0  = allocH((size_t)(n0 + 1) * 32);
  half_t* d1f = allocH((size_t)(n1 + 1) * 64);
  half_t* t1  = allocH((size_t)(n1 + 1) * 64);
  half_t* e1  = allocH((size_t)(n1 + 1) * 64);
  half_t* d2f = allocH((size_t)(n2 + 1) * 128);
  half_t* t2  = allocH((size_t)(n2 + 1) * 128);
  half_t* e2  = allocH((size_t)(n2 + 1) * 128);
  half_t* d3f = allocH((size_t)(n3 + 1) * 256);
  half_t* t3  = allocH((size_t)(n3 + 1) * 256);
  half_t* e3  = allocH((size_t)(n3 + 1) * 256);
  half_t* u3  = allocH((size_t)(n2 + 1) * 128);
  half_t* c3  = allocH((size_t)(n2 + 1) * 256);
  half_t* dd3 = allocH((size_t)(n2 + 1) * 128);
  half_t* u2  = allocH((size_t)(n1 + 1) * 64);
  half_t* c2  = allocH((size_t)(n1 + 1) * 128);
  half_t* dd2 = allocH((size_t)(n1 + 1) * 64);
  half_t* u1  = allocH((size_t)(n0 + 1) * 32);
  half_t* c1  = allocH((size_t)(n0 + 1) * 64);
  half_t* dd1 = allocH((size_t)(n0 + 1) * 32);
  (void)ws_size;

  // zero the pad rows of every conv-input buffer (one tiny launch each)
  struct PadRow { half_t* p; int n; int c; };
  const PadRow pads[] = {
    {x0, n0, 32}, {t0, n0, 32}, {e0, n0, 32},
    {d1f, n1, 64}, {t1, n1, 64}, {e1, n1, 64},
    {d2f, n2, 128}, {t2, n2, 128}, {e2, n2, 128},
    {d3f, n3, 256}, {t3, n3, 256}, {e3, n3, 256},
    {u3, n2, 128}, {dd3, n2, 128}, {u2, n1, 64}, {dd2, n1, 64}, {u1, n0, 32},
  };
  for (const PadRow& pr : pads)
    zero_f16<<<1, 256, 0, stream>>>(pr.p + (size_t)pr.n * pr.c, pr.c);

  // ---- voxelize: segment mean -> f16 padded rows ----
  zero_f32<<<(unsigned)((n0 + 255) / 256), 256, 0, stream>>>(cnt, n0);
  {
    long nv = (long)n0 * 9;
    zero_f32<<<(unsigned)((nv + 255) / 256), 256, 0, stream>>>(voxf, nv);
  }
  vox_accum<<<(npts + 127) / 128, 128, 0, stream>>>(feat, pvid, voxf, cnt, npts);
  {
    long nv = (long)(n0 + 1) * 32;
    vox_to_f16<<<(unsigned)((nv + 255) / 256), 256, 0, stream>>>(voxf, cnt, vox, n0);
  }

  auto G = [&](int i) { return (const float*)d_in[kLayers[i].g]; };
  auto B = [&](int i) { return (const float*)d_in[kLayers[i].b]; };

  // ---- encoder ----
  launch_spconv(stream, vox, nbr0, wt16[0], G(0), B(0), x0, n0, n0, 32, 32);
  launch_spconv(stream, x0,  nbr0, wt16[1], G(1), B(1), t0, n0, n0, 32, 32);
  launch_spconv(stream, t0,  nbr0, wt16[2], G(2), B(2), e0, n0, n0, 32, 32);
  launch_spconv(stream, e0,  dn1,  wt16[3], G(3), B(3), d1f, n0, n1, 32, 64);
  launch_spconv(stream, d1f, nbr1, wt16[4], G(4), B(4), t1, n1, n1, 64, 64);
  launch_spconv(stream, t1,  nbr1, wt16[5], G(5), B(5), e1, n1, n1, 64, 64);
  launch_spconv(stream, e1,  dn2,  wt16[6], G(6), B(6), d2f, n1, n2, 64, 128);
  launch_spconv(stream, d2f, nbr2, wt16[7], G(7), B(7), t2, n2, n2, 128, 128);
  launch_spconv(stream, t2,  nbr2, wt16[8], G(8), B(8), e2, n2, n2, 128, 128);
  launch_spconv(stream, e2,  dn3,  wt16[9], G(9), B(9), d3f, n2, n3, 128, 256);
  launch_spconv(stream, d3f, nbr3, wt16[10], G(10), B(10), t3, n3, n3, 256, 256);
  launch_spconv(stream, t3,  nbr3, wt16[11], G(11), B(11), e3, n3, n3, 256, 256);

  // ---- decoder with skip concat (concat covers pad row too) ----
  launch_spconv(stream, e3, iv3, wt16[12], G(12), B(12), u3, n3, n2, 256, 128);
  concat2_f16<<<(unsigned)(((long)(n2 + 1) * 256 + 255) / 256), 256, 0, stream>>>(
      u3, 128, e2, 128, c3, n2 + 1);
  launch_spconv(stream, c3, nbr2, wt16[13], G(13), B(13), dd3, n2, n2, 256, 128);

  launch_spconv(stream, dd3, iv2, wt16[14], G(14), B(14), u2, n2, n1, 128, 64);
  concat2_f16<<<(unsigned)(((long)(n1 + 1) * 128 + 255) / 256), 256, 0, stream>>>(
      u2, 64, e1, 64, c2, n1 + 1);
  launch_spconv(stream, c2, nbr1, wt16[15], G(15), B(15), dd2, n1, n1, 128, 64);

  launch_spconv(stream, dd2, iv1, wt16[16], G(16), B(16), u1, n1, n0, 64, 32);
  concat2_f16<<<(unsigned)(((long)(n0 + 1) * 64 + 255) / 256), 256, 0, stream>>>(
      u1, 32, e0, 32, c1, n0 + 1);
  launch_spconv(stream, c1, nbr0, wt16[17], G(17), B(17), dd1, n0, n0, 64, 32);

  // ---- devoxelize + heads ----
  float* out_pf  = (float*)d_out;
  float* out_sem = out_pf + (size_t)npts * 32;
  float* out_off = out_sem + (size_t)npts * 2;
  heads_kernel<<<(npts + 127) / 128, 128, 0, stream>>>(
      dd1, pvid,
      (const float*)d_in[P_SEM_W1], (const float*)d_in[P_SEM_W2],
      (const float*)d_in[P_SEM_B1], (const float*)d_in[P_SEM_B2],
      (const float*)d_in[P_OFF_W1], (const float*)d_in[P_OFF_W2],
      (const float*)d_in[P_OFF_B1], (const float*)d_in[P_OFF_B2],
      out_pf, out_sem, out_off, npts);
  (void)n_in; (void)out_size;
}
